// CRF_36910948942223
// MI455X (gfx1250) — compile-verified, matrix operand order checked by reading, not run on hardware
//
#include <hip/hip_runtime.h>

typedef __attribute__((ext_vector_type(16))) _Float16 v16h;
typedef __attribute__((ext_vector_type(8)))  float    v8f;

#define NTAGS     64
#define TSTEPS    512
#define BATCH     1024
#define START_TAG 62
#define STOP_TAG  63
#define NEG_INF_F (-10000.0f)
#define LOG2E_F   1.4426950408889634f
#define LN2_F     0.6931471805599453f
#define PF_DIST   12

// ---------------------------------------------------------------------------
// Kernel 1: E[i][j] = exp(trans[i][j]) as f16 (forbidden transitions -> 0).
// ---------------------------------------------------------------------------
__global__ void crf_prep_expE(const float* __restrict__ trans,
                              _Float16* __restrict__ Ebuf) {
    int i = blockIdx.x * blockDim.x + threadIdx.x;
    if (i < NTAGS * NTAGS) Ebuf[i] = (_Float16)__expf(trans[i]);
}

// ---------------------------------------------------------------------------
// Kernel 2: gold path score, one thread per batch (sequential tag scan).
// ---------------------------------------------------------------------------
__global__ void crf_gold(const float* __restrict__ emis,
                         const float* __restrict__ trans,
                         const int*   __restrict__ tags,
                         const int*   __restrict__ mask,
                         float* __restrict__ gold) {
    int b = blockIdx.x * blockDim.x + threadIdx.x;
    if (b >= BATCH) return;
    float score = 0.0f;
    int prev = START_TAG;
    const int*   tg = tags + (size_t)b * TSTEPS;
    const int*   mk = mask + (size_t)b * TSTEPS;
    const float* em = emis + (size_t)b * TSTEPS * NTAGS;
    for (int t = 0; t < TSTEPS; ++t) {
        int tag = tg[t];
        if (mk[t]) {
            score += em[(size_t)t * NTAGS + tag] + trans[prev * NTAGS + tag];
            prev = tag;
        }
    }
    gold[b] = score + trans[prev * NTAGS + STOP_TAG];
}

// ---------------------------------------------------------------------------
// Kernel 3: forward scan. 4 waves (128 threads) per 16-batch group.
//   phase A (128 threads): s = rowmax(alpha); P = exp2((alpha-s)*log2e) -> LDS f16
//   phase B (per wave w):  Q[:,16w:16w+16] = P(16x64) @ E[:,16w:16w+16]
//                          via 2x v_wmma_f32_16x16x32_f16
//   phase C (per wave w):  alpha'[:,tile w] = select(mask, s+ln(Q)+emit, alpha)
// ---------------------------------------------------------------------------
__launch_bounds__(128)
__global__ void crf_forward(const float* __restrict__ emis,
                            const float* __restrict__ trans,
                            const int*   __restrict__ mask,
                            const _Float16* __restrict__ Ebuf,
                            float* __restrict__ logZ) {
    __shared__ float    alpha[16 * 65];   // padded rows vs bank conflicts
    __shared__ _Float16 pbuf[16 * 68];    // padded rows
    __shared__ __attribute__((aligned(16))) float srow[16];
    __shared__ __attribute__((aligned(16))) int   mrow[16];
    __shared__ float    pmax[128];

    const int tid  = threadIdx.x;   // 0..127
    const int lane = tid & 31;      // lane in wave
    const int w    = tid >> 5;      // wave id = j-tile id (0..3)
    const int lo   = lane & 15;
    const int hi   = lane >> 4;     // 0 or 1
    const int row  = tid & 15;      // phase-A row
    const int q    = tid >> 4;      // phase-A quarter-of-row (0..7), 8 states each
    const int b0   = blockIdx.x * 16;

    // Preload this wave's constant B operand (E) tile, 16-bit 32x16 WMMA layout.
    // Half h of v16h: v=h/2, parity=h&1 -> K_local = (v<4 ? 2v : 8+2v)+parity+8*hi
    v16h Bt0, Bt1;
    #pragma unroll
    for (int h = 0; h < 16; ++h) {
        int v = h >> 1, par = h & 1;
        int kl = ((v < 4) ? 2 * v : 8 + 2 * v) + par + 8 * hi;
        Bt0[h] = Ebuf[kl * NTAGS + w * 16 + lo];
        Bt1[h] = Ebuf[(32 + kl) * NTAGS + w * 16 + lo];
    }

    // Per-row emissions base pointers for this wave's tile (D-layout row
    // m = r + 8*hi, column w*16 + lo). t advances by NTAGS floats.
    const float* embase[8];
    #pragma unroll
    for (int r = 0; r < 8; ++r)
        embase[r] = emis + (size_t)(b0 + r + 8 * hi) * TSTEPS * NTAGS + w * 16 + lo;
    const int* mbase = mask + (size_t)(b0 + row) * TSTEPS;   // used by q==0 threads

    // alpha0: NEG_INF everywhere except START.
    for (int k = tid; k < 16 * 64; k += 128) {
        int m = k >> 6, n = k & 63;
        alpha[m * 65 + n] = (n == START_TAG) ? 0.0f : NEG_INF_F;
    }
    __syncthreads();

    for (int t = 0; t < TSTEPS; ++t) {
        // ---- issue this step's emissions loads early (overlap w/ A+B) ----
        float ev[8];
        #pragma unroll
        for (int r = 0; r < 8; ++r) {
            const float* pr = embase[r] + (size_t)t * NTAGS;
            ev[r] = *pr;
            if (t + PF_DIST < TSTEPS)
                __builtin_prefetch(pr + (size_t)PF_DIST * NTAGS, 0, 1);
        }

        // ---- phase A: row max + exp2-normalize into f16 P buffer ----
        {
            const float* ar = &alpha[row * 65 + q * 8];
            float mx = NEG_INF_F;
            #pragma unroll
            for (int k = 0; k < 8; ++k) mx = fmaxf(mx, ar[k]);
            pmax[tid] = mx;
            __syncthreads();
            float s = NEG_INF_F;
            #pragma unroll
            for (int j = 0; j < 8; ++j) s = fmaxf(s, pmax[j * 16 + row]);
            if (q == 0) {
                srow[row] = s;
                mrow[row] = mbase[t];
            }
            _Float16* pr = &pbuf[row * 68 + q * 8];
            #pragma unroll
            for (int k = 0; k < 8; ++k)
                pr[k] = (_Float16)__builtin_amdgcn_exp2f((ar[k] - s) * LOG2E_F);
            __syncthreads();
        }

        // ---- phase B: load A tiles (16-bit 16x32 layout) and do WMMA ----
        v16h A0, A1;
        #pragma unroll
        for (int h = 0; h < 16; ++h) {
            int v = h >> 1, par = h & 1;
            int kl = ((v < 4) ? 2 * v : 8 + 2 * v) + par + 8 * hi;
            A0[h] = pbuf[lo * 68 + kl];
            A1[h] = pbuf[lo * 68 + 32 + kl];
        }
        v8f acc = {};
        acc = __builtin_amdgcn_wmma_f32_16x16x32_f16(false, A0, false, Bt0,
                                                     (short)0, acc, false, false);
        acc = __builtin_amdgcn_wmma_f32_16x16x32_f16(false, A1, false, Bt1,
                                                     (short)0, acc, false, false);

        // ---- phase C: branchless masked update (this lane owns its 8 cells)
        float sv[8], aold[8];
        int   mv[8];
        #pragma unroll
        for (int r = 0; r < 8; ++r) {
            sv[r]   = srow[8 * hi + r];
            mv[r]   = mrow[8 * hi + r];
            aold[r] = alpha[(r + 8 * hi) * 65 + w * 16 + lo];
        }
        float nv[8];
        #pragma unroll
        for (int r = 0; r < 8; ++r) {
            float lq = __builtin_amdgcn_logf(acc[r]) * LN2_F;   // ln(0) -> -inf
            nv[r] = fmaxf(sv[r] + lq, NEG_INF_F) + ev[r];
        }
        #pragma unroll
        for (int r = 0; r < 8; ++r)
            alpha[(r + 8 * hi) * 65 + w * 16 + lo] = mv[r] ? nv[r] : aold[r];
        __syncthreads();
    }

    // ---- finalize: logZ[b] = logsumexp_k(alpha[b,k] + trans[k,STOP]) ----
    {
        const float* ar = &alpha[row * 65 + q * 8];
        float mx = NEG_INF_F;
        #pragma unroll
        for (int k = 0; k < 8; ++k)
            mx = fmaxf(mx, ar[k] + trans[(q * 8 + k) * NTAGS + STOP_TAG]);
        pmax[tid] = mx;
        __syncthreads();
        float s = NEG_INF_F;
        #pragma unroll
        for (int j = 0; j < 8; ++j) s = fmaxf(s, pmax[j * 16 + row]);
        float sum = 0.0f;
        #pragma unroll
        for (int k = 0; k < 8; ++k)
            sum += __builtin_amdgcn_exp2f(
                (ar[k] + trans[(q * 8 + k) * NTAGS + STOP_TAG] - s) * LOG2E_F);
        __syncthreads();
        pmax[tid] = sum;
        __syncthreads();
        if (q == 0) {
            float tot = 0.0f;
            #pragma unroll
            for (int j = 0; j < 8; ++j) tot += pmax[j * 16 + row];
            logZ[b0 + row] = s + __builtin_amdgcn_logf(tot) * LN2_F;
        }
    }
}

// ---------------------------------------------------------------------------
// Kernel 4: out = mean(logZ - gold)
// ---------------------------------------------------------------------------
__global__ void crf_reduce(const float* __restrict__ logZ,
                           const float* __restrict__ gold,
                           float* __restrict__ out) {
    __shared__ float sh[256];
    int tid = threadIdx.x;
    float s = 0.0f;
    for (int b = tid; b < BATCH; b += 256) s += logZ[b] - gold[b];
    sh[tid] = s;
    __syncthreads();
    for (int w = 128; w > 0; w >>= 1) {
        if (tid < w) sh[tid] += sh[tid + w];
        __syncthreads();
    }
    if (tid == 0) out[0] = sh[0] / (float)BATCH;
}

// ---------------------------------------------------------------------------
extern "C" void kernel_launch(void* const* d_in, const int* in_sizes, int n_in,
                              void* d_out, int out_size, void* d_ws, size_t ws_size,
                              hipStream_t stream) {
    const float* emis  = (const float*)d_in[0];  // (B,T,C) f32
    const float* trans = (const float*)d_in[1];  // (C,C)   f32
    const int*   tags  = (const int*)d_in[2];    // (B,T)   int
    const int*   mask  = (const int*)d_in[3];    // (B,T)   int

    char* ws = (char*)d_ws;
    _Float16* Ebuf = (_Float16*)ws;                        // 64*64*2 = 8 KB
    float*    logZ = (float*)(ws + 8192);                  // 4 KB
    float*    gold = (float*)(ws + 8192 + 4096);           // 4 KB

    crf_prep_expE<<<(NTAGS * NTAGS + 255) / 256, 256, 0, stream>>>(trans, Ebuf);
    crf_gold<<<BATCH / 256, 256, 0, stream>>>(emis, trans, tags, mask, gold);
    crf_forward<<<BATCH / 16, 128, 0, stream>>>(emis, trans, mask, Ebuf, logZ);
    crf_reduce<<<1, 256, 0, stream>>>(logZ, gold, (float*)d_out);
}
